// VariableSelectionNetwork_29180007809522
// MI455X (gfx1250) — compile-verified, matrix operand order checked by reading, not run on hardware
//
#include <hip/hip_runtime.h>
#include <stdint.h>

#define NF 16
#define NU 256
#define TWO_U 512
#define NBT 32768   // 64*512
#define ROWS 32     // rows per workgroup (two 16-row WMMA M-tiles)
#define LN_EPS 1e-6f

typedef __attribute__((ext_vector_type(16))) __bf16 v16bf;
typedef __attribute__((ext_vector_type(8)))  float  v8f;
typedef __attribute__((ext_vector_type(4)))  unsigned int u32x4;

union Frag { u32x4 q[2]; v16bf v; };

__device__ __forceinline__ unsigned short f2bf(float f) {
    union { float f; uint32_t u; } c; c.f = f;
    uint32_t u = c.u;
    uint32_t r = u + 0x7FFFu + ((u >> 16) & 1u);      // round-to-nearest-even
    if ((u & 0x7F800000u) == 0x7F800000u) r = u;      // inf/nan: truncate
    return (unsigned short)(r >> 16);
}

__device__ __forceinline__ float eluf(float z)  { return z > 0.f ? z : (__expf(z) - 1.f); }
__device__ __forceinline__ float sigmf(float z) { return 1.f / (1.f + __expf(-z)); }

// ---------------------------------------------------------------------------
// Kernel 1: fold the rank-1 per-feature projection into the GRN weights.
//   P[i,v]  = sum_u fp_w[i,u]*fg_proj[i,u,v]     Pb[i,v] = sum_u fp_b[i,u]*fg_proj[i,u,v]
//   W1[i,v] = sum_u fp_w[i,u]*fg_d1w[i,u,v]      B1[i,v] = sum_u fp_b[i,u]*fg_d1w[i,u,v] + fg_d1b[i,v]
// ---------------------------------------------------------------------------
__global__ void vsn_precompute(const float* __restrict__ fp_w, const float* __restrict__ fp_b,
                               const float* __restrict__ fg_proj, const float* __restrict__ fg_d1w,
                               const float* __restrict__ fg_d1b,
                               float* __restrict__ W1, float* __restrict__ B1,
                               float* __restrict__ P,  float* __restrict__ Pb) {
    int t = blockIdx.x * blockDim.x + threadIdx.x;   // NF*NU threads
    int i = t >> 8;
    int v = t & 255;
    const float* wrow = fp_w + i * NU;
    const float* brow = fp_b + i * NU;
    const float* gp = fg_proj + (size_t)i * NU * NU + v;
    const float* g1 = fg_d1w  + (size_t)i * NU * NU + v;
    float p = 0.f, pb = 0.f, w1 = 0.f, b1 = 0.f;
    for (int u = 0; u < NU; ++u) {
        float fw = wrow[u], fb = brow[u];
        float a = gp[(size_t)u * NU];
        float b = g1[(size_t)u * NU];
        p  += fw * a;  pb += fb * a;
        w1 += fw * b;  b1 += fb * b;
    }
    W1[t] = w1; B1[t] = b1 + fg_d1b[t]; P[t] = p; Pb[t] = pb;
}

// ---------------------------------------------------------------------------
// Kernel 2: repack fg_d2w [F,256,512] fp32 into bf16 WMMA B-fragment blobs.
// Blob (i,kk,nn): 32 lanes x 16 bf16 (1024B contiguous). Lane l<16 holds
// column N=nn*16+l, K = kk*32 + {0..7,16..23}; lane l>=16: same N set with
// K offset +8 (mirrors the documented 16-bit A-matrix lane layout).
// ---------------------------------------------------------------------------
__global__ void vsn_pack_d2w(const float* __restrict__ fg_d2w, unsigned short* __restrict__ pack) {
    int t = blockIdx.x * blockDim.x + threadIdx.x;   // 2^21 threads
    int e  = t & 15;
    int l  = (t >> 4) & 31;
    int nn = (t >> 9) & 31;
    int kk = (t >> 14) & 7;
    int i  = t >> 17;
    int half = l >> 4;
    int n = nn * 16 + (l & 15);
    int k = kk * 32 + half * 8 + (e < 8 ? e : 8 + e);
    pack[t] = f2bf(fg_d2w[(size_t)i * NU * TWO_U + (size_t)k * TWO_U + n]);
}

// ---------------------------------------------------------------------------
// Kernel 3: fused VSN. One WG = 32 (b,t) rows, all 16 features, U=256 outputs.
// Two 16-row M-tiles share every loaded B fragment (2 WMMAs per B load).
// ---------------------------------------------------------------------------
__global__ __launch_bounds__(256)
void vsn_main(const float* __restrict__ x,
              const float* __restrict__ W1, const float* __restrict__ B1,
              const float* __restrict__ P,  const float* __restrict__ Pb,
              const unsigned short* __restrict__ packB,
              const float* __restrict__ fg_d2b, const float* __restrict__ fg_g,
              const float* __restrict__ fg_bt,
              const float* __restrict__ cg_proj, const float* __restrict__ cg_d1w,
              const float* __restrict__ cg_d1b,  const float* __restrict__ cg_d2w,
              const float* __restrict__ cg_d2b,  const float* __restrict__ cg_g,
              const float* __restrict__ cg_bt,
              float* __restrict__ out, float* __restrict__ outw)
{
    constexpr int ASTR = NU + 8;      // bf16 row stride (pad -> conflict-free b128 reads)
    constexpr int HSTR = TWO_U + 4;   // f32 row stride (pad -> conflict-free tile stores)
    __shared__ float xs[ROWS][NF];
    __shared__ float wsm[ROWS][NF];
    __shared__ __align__(16) unsigned short asm_[ROWS * ASTR];
    __shared__ float hs[16 * HSTR];               // one 16-row M-tile at a time
    __shared__ float acc[ROWS][NU];

    const int tid  = threadIdx.x;
    const int lane = tid & 31;
    const int wave = tid >> 5;
    const int half = lane >> 4;
    const int l15  = lane & 15;
    const int rowbase = blockIdx.x * ROWS;

    { // x tile: 32 rows x 16 features (512 floats, 2 per thread)
        int idx = tid;
        xs[idx >> 4][idx & 15] = x[(size_t)rowbase * NF + idx];
        idx = tid + 256;
        xs[idx >> 4][idx & 15] = x[(size_t)rowbase * NF + idx];
    }
#pragma unroll
    for (int m = 0; m < ROWS; ++m) acc[m][tid] = 0.f;
    __syncthreads();

    // ---- combined GRN -> softmax feature weights (waves 0-1, one row/lane) ----
    if (wave < 2 && lane < 16) {
        int r = wave * 16 + lane;
        float xr[NF], rc[NF], ac[NF], z[NF];
        for (int j = 0; j < NF; ++j) xr[j] = xs[r][j];
        for (int f = 0; f < NF; ++f) {
            float s0 = 0.f, s1 = cg_d1b[f];
            for (int j = 0; j < NF; ++j) {
                s0 += xr[j] * cg_proj[j * NF + f];
                s1 += xr[j] * cg_d1w[j * NF + f];
            }
            rc[f] = s0;
            ac[f] = eluf(s1);
        }
        for (int f = 0; f < NF; ++f) {
            float t1 = cg_d2b[f], t2 = cg_d2b[NF + f];
            for (int j = 0; j < NF; ++j) {
                t1 += ac[j] * cg_d2w[j * 2 * NF + f];
                t2 += ac[j] * cg_d2w[j * 2 * NF + NF + f];
            }
            z[f] = rc[f] + t1 * sigmf(t2);
        }
        float mu = 0.f;
        for (int f = 0; f < NF; ++f) mu += z[f];
        mu *= (1.f / NF);
        float var = 0.f;
        for (int f = 0; f < NF; ++f) { float d = z[f] - mu; var += d * d; }
        var *= (1.f / NF);
        float inv = rsqrtf(var + LN_EPS);
        float mx = -1e30f;
        for (int f = 0; f < NF; ++f) {
            z[f] = (z[f] - mu) * inv * cg_g[f] + cg_bt[f];
            mx = fmaxf(mx, z[f]);
        }
        float se = 0.f;
        for (int f = 0; f < NF; ++f) { z[f] = __expf(z[f] - mx); se += z[f]; }
        float rs = 1.f / se;
        for (int f = 0; f < NF; ++f) wsm[r][f] = z[f] * rs;
    }
    __syncthreads();

    // ---- per-feature heavy path ----
    for (int i = 0; i < NF; ++i) {
        { // a = elu(x_i * W1 + B1) -> bf16 LDS (thread = column, loop 32 rows)
            float w1 = W1[i * NU + tid];
            float b1 = B1[i * NU + tid];
#pragma unroll
            for (int m = 0; m < ROWS; ++m)
                asm_[m * ASTR + tid] = f2bf(eluf(xs[m][i] * w1 + b1));
        }
        __syncthreads();

        // h[32x512] = a[32x256] @ B_i[256x512]; each B frag feeds 2 WMMAs
        v8f cacc[2][4];
#pragma unroll
        for (int mt = 0; mt < 2; ++mt)
#pragma unroll
            for (int j = 0; j < 4; ++j) cacc[mt][j] = (v8f){};
        const u32x4* bp = reinterpret_cast<const u32x4*>(packB);
#pragma unroll
        for (int kk = 0; kk < 8; ++kk) {
            Frag a0, a1;
            const unsigned short* ar0 = asm_ + l15 * ASTR + kk * 32 + half * 8;
            const unsigned short* ar1 = ar0 + 16 * ASTR;
            a0.q[0] = *reinterpret_cast<const u32x4*>(ar0);
            a0.q[1] = *reinterpret_cast<const u32x4*>(ar0 + 16);
            a1.q[0] = *reinterpret_cast<const u32x4*>(ar1);
            a1.q[1] = *reinterpret_cast<const u32x4*>(ar1 + 16);
            size_t base = ((size_t)((i * 8 + kk) * 32 + wave * 4)) * 64 + (size_t)lane * 2;
#pragma unroll
            for (int j = 0; j < 4; ++j) {
                Frag b;
                b.q[0] = bp[base + j * 64];
                b.q[1] = bp[base + j * 64 + 1];
                cacc[0][j] = __builtin_amdgcn_wmma_f32_16x16x32_bf16(
                    false, a0.v, false, b.v, (short)0, cacc[0][j], false, false);
                cacc[1][j] = __builtin_amdgcn_wmma_f32_16x16x32_bf16(
                    false, a1.v, false, b.v, (short)0, cacc[1][j], false, false);
            }
        }

        // post-process the two M-tiles sequentially through the 16-row hs buffer
#pragma unroll
        for (int mt = 0; mt < 2; ++mt) {
            { // C/D tile layout: VGPR r, lanes 0-15 -> M=r, lanes 16-31 -> M=r+8
                int mrow = half * 8;
#pragma unroll
                for (int j = 0; j < 4; ++j) {
                    int col = (wave * 4 + j) * 16 + l15;
#pragma unroll
                    for (int r = 0; r < 8; ++r)
                        hs[(mrow + r) * HSTR + col] = cacc[mt][j][r];
                }
            }
            __syncthreads();

            // gated residual + layernorm + weighted accumulation (2 rows/wave)
#pragma unroll
            for (int rr = 0; rr < 2; ++rr) {
                int lrow = 2 * wave + rr;           // row within hs tile
                int row  = mt * 16 + lrow;          // logical row within WG
                float xi = xs[row][i];
                float wt = wsm[row][i];
                float zv[8];
                float s = 0.f, s2 = 0.f;
#pragma unroll
                for (int j = 0; j < 8; ++j) {
                    int v = lane + 32 * j;
                    float h1 = hs[lrow * HSTR + v]      + fg_d2b[i * TWO_U + v];
                    float h2 = hs[lrow * HSTR + NU + v] + fg_d2b[i * TWO_U + NU + v];
                    float rv = xi * P[i * NU + v] + Pb[i * NU + v];
                    float z  = rv + h1 * sigmf(h2);
                    zv[j] = z;
                    s += z; s2 += z * z;
                }
#pragma unroll
                for (int off = 16; off >= 1; off >>= 1) {
                    s  += __shfl_xor(s,  off, 32);
                    s2 += __shfl_xor(s2, off, 32);
                }
                float mu  = s * (1.f / NU);
                float var = s2 * (1.f / NU) - mu * mu;
                float inv = rsqrtf(var + LN_EPS);
#pragma unroll
                for (int j = 0; j < 8; ++j) {
                    int v = lane + 32 * j;
                    float zn = (zv[j] - mu) * inv * fg_g[i * NU + v] + fg_bt[i * NU + v];
                    acc[row][v] += zn * wt;
                }
            }
            __syncthreads();
        }
    }

    // ---- outputs: out [BT,256] then w [BT,16] ----
#pragma unroll
    for (int m = 0; m < ROWS; ++m)
        out[(size_t)(rowbase + m) * NU + tid] = acc[m][tid];
    {
        outw[(size_t)rowbase * NF + tid]       = wsm[tid >> 4][tid & 15];
        outw[(size_t)rowbase * NF + tid + 256] = wsm[(tid + 256) >> 4][tid & 15];
    }
}

extern "C" void kernel_launch(void* const* d_in, const int* in_sizes, int n_in,
                              void* d_out, int out_size, void* d_ws, size_t ws_size,
                              hipStream_t stream) {
    const float* x       = (const float*)d_in[0];
    const float* fp_w    = (const float*)d_in[1];
    const float* fp_b    = (const float*)d_in[2];
    const float* fg_proj = (const float*)d_in[3];
    const float* fg_d1w  = (const float*)d_in[4];
    const float* fg_d1b  = (const float*)d_in[5];
    const float* fg_d2w  = (const float*)d_in[6];
    const float* fg_d2b  = (const float*)d_in[7];
    const float* fg_g    = (const float*)d_in[8];
    const float* fg_bt   = (const float*)d_in[9];
    const float* cg_proj = (const float*)d_in[10];
    const float* cg_d1w  = (const float*)d_in[11];
    const float* cg_d1b  = (const float*)d_in[12];
    const float* cg_d2w  = (const float*)d_in[13];
    const float* cg_d2b  = (const float*)d_in[14];
    const float* cg_g    = (const float*)d_in[15];
    const float* cg_bt   = (const float*)d_in[16];

    // workspace: 4 fused per-feature vectors (fp32) + bf16 fragment-packed fg_d2w
    float* W1 = (float*)d_ws;
    float* B1 = W1 + NF * NU;
    float* P  = B1 + NF * NU;
    float* Pb = P  + NF * NU;
    unsigned short* packB = (unsigned short*)(Pb + NF * NU);   // 4 MB

    float* out  = (float*)d_out;
    float* outw = out + (size_t)NBT * NU;

    vsn_precompute<<<(NF * NU) / 256, 256, 0, stream>>>(fp_w, fp_b, fg_proj, fg_d1w, fg_d1b,
                                                        W1, B1, P, Pb);
    vsn_pack_d2w<<<(NF * NU * TWO_U) / 256, 256, 0, stream>>>(fg_d2w, packB);
    vsn_main<<<NBT / ROWS, 256, 0, stream>>>(x, W1, B1, P, Pb, packB,
                                             fg_d2b, fg_g, fg_bt,
                                             cg_proj, cg_d1w, cg_d1b, cg_d2w, cg_d2b, cg_g, cg_bt,
                                             out, outw);
}
